// MaskedAttentionHead_28698971471981
// MI455X (gfx1250) — matrix operand order repaired and checked by reading, not compiled, not run on hardware
//
#include <hip/hip_runtime.h>
#include <hip/hip_bf16.h>

typedef __attribute__((ext_vector_type(16))) __bf16 v16bf;
typedef __attribute__((ext_vector_type(8)))  __bf16 v8bf;
typedef __attribute__((ext_vector_type(8)))  float  v8f;

#define BATCH 32
#define SEQ   1024
#define EMB   768
#define HD    64
#define NIMG  49

// ---------------------------------------------------------------------------
// WMMA wrapper: D = A(16x32 bf16) x B(32x16 bf16) + C(16x16 f32)
// ---------------------------------------------------------------------------
__device__ inline v8f wmma_bf16(v16bf a, v16bf b, v8f c) {
    return __builtin_amdgcn_wmma_f32_16x16x32_bf16(
        /*neg_a=*/false, a, /*neg_b=*/false, b,
        /*c_mod=*/(short)0, c, /*reuse_a=*/false, /*reuse_b=*/false);
}

// A-fragment (16x32, bf16) from a row-major bf16 matrix with row stride ld.
__device__ inline v16bf load_a_bf16(const __bf16* p, int ld, int rbase, int k0, int lane) {
    const int m  = lane & 15;
    const int kh = lane >> 4;
    const __bf16* r = p + (size_t)(rbase + m) * (size_t)ld + k0;
    v8bf lo = *(const v8bf*)(r + kh * 8);
    v8bf hi = *(const v8bf*)(r + 16 + kh * 8);
    v16bf o;
#pragma unroll
    for (int e = 0; e < 8; ++e) { o[e] = lo[e]; o[e + 8] = hi[e]; }
    return o;
}

// B-fragment (32x16, bf16): K contiguous per column (column stride elements).
__device__ inline v16bf load_b_bf16(const __bf16* p, int colstride, int nbase, int k0, int lane) {
    const int n  = lane & 15;
    const int kh = lane >> 4;
    const __bf16* c = p + (size_t)(nbase + n) * (size_t)colstride + k0;
    v8bf lo = *(const v8bf*)(c + kh * 8);
    v8bf hi = *(const v8bf*)(c + 16 + kh * 8);
    v16bf o;
#pragma unroll
    for (int e = 0; e < 8; ++e) { o[e] = lo[e]; o[e + 8] = hi[e]; }
    return o;
}

// A-fragment from an f32 row-major matrix, converted to bf16 on the fly.
__device__ inline v16bf load_a_f32cvt(const float* p, int ld, int rbase, int k0, int lane) {
    const int m  = lane & 15;
    const int kh = lane >> 4;
    const float* r = p + (size_t)(rbase + m) * (size_t)ld + k0;
    const float4* q0 = (const float4*)(r + kh * 8);
    const float4* q1 = (const float4*)(r + 16 + kh * 8);
    float4 a0 = q0[0], a1 = q0[1], b0 = q1[0], b1 = q1[1];
    v16bf o;
    o[0]  = (__bf16)a0.x; o[1]  = (__bf16)a0.y; o[2]  = (__bf16)a0.z; o[3]  = (__bf16)a0.w;
    o[4]  = (__bf16)a1.x; o[5]  = (__bf16)a1.y; o[6]  = (__bf16)a1.z; o[7]  = (__bf16)a1.w;
    o[8]  = (__bf16)b0.x; o[9]  = (__bf16)b0.y; o[10] = (__bf16)b0.z; o[11] = (__bf16)b0.w;
    o[12] = (__bf16)b1.x; o[13] = (__bf16)b1.y; o[14] = (__bf16)b1.z; o[15] = (__bf16)b1.w;
    return o;
}

// ---------------------------------------------------------------------------
// Kernel 0: transpose + convert weights.  W[768][64] f32 -> Wt[mat][64][768] bf16
// ---------------------------------------------------------------------------
__global__ __launch_bounds__(256) void prep_weights(const float* __restrict__ Wq,
                                                    const float* __restrict__ Wk,
                                                    const float* __restrict__ Wv,
                                                    __bf16* __restrict__ Wt) {
    int idx = blockIdx.x * 256 + threadIdx.x;
    const int per = EMB * HD;
    if (idx >= 3 * per) return;
    int mat = idx / per;
    int rem = idx - mat * per;
    int k = rem >> 6;       // 0..767
    int c = rem & 63;       // 0..63
    const float* W = (mat == 0) ? Wq : (mat == 1) ? Wk : Wv;
    Wt[(size_t)mat * per + (size_t)c * EMB + k] = (__bf16)W[rem];
}

// ---------------------------------------------------------------------------
// Kernel 1: fused QKV projection.
// Grid 256 WGs x 256 threads (8 waves). WG covers 128 rows x 192 cols.
// Each wave: 2 row-tiles x 6 col-tiles (12 accumulators). Per k-step only
// 6 B-fragments are loaded and each feeds TWO WMMAs, halving the
// load-wait : wmma ratio. min-waves hint = 1 frees VGPRs so the batched
// fragment loads can stay hoisted ahead of the WMMA block.
// ---------------------------------------------------------------------------
__global__ __launch_bounds__(256, 1) void proj_kernel(const float* __restrict__ X,
                                                      const __bf16* __restrict__ Wt,
                                                      const float* __restrict__ bq,
                                                      const float* __restrict__ bk,
                                                      const float* __restrict__ bv,
                                                      __bf16* __restrict__ Qbf,
                                                      __bf16* __restrict__ Kbf,
                                                      __bf16* __restrict__ Vt) {
    const int lane   = threadIdx.x & 31;
    const int wave   = threadIdx.x >> 5;
    const int rowgrp = wave >> 1;          // 0..3
    const int colgrp = wave & 1;           // 0..1
    const int rowbase = blockIdx.x * 128 + rowgrp * 32;  // two 16-row tiles
    const int tbase   = colgrp * 6;                      // six 16-col tiles

    v8f acc0[6] = {};   // rows [rowbase, rowbase+16)
    v8f acc1[6] = {};   // rows [rowbase+16, rowbase+32)

    v16bf a0 = load_a_f32cvt(X, EMB, rowbase,      0, lane);
    v16bf a1 = load_a_f32cvt(X, EMB, rowbase + 16, 0, lane);

    for (int kk = 0; kk < EMB; kk += 32) {
        // batch the 6 weight-fragment loads for this k-step
        v16bf bfr[6];
#pragma unroll
        for (int u = 0; u < 6; ++u) {
            const int t = tbase + u;
            const __bf16* w = Wt + (size_t)(t >> 2) * (EMB * HD);
            bfr[u] = load_b_bf16(w, EMB, (t & 3) * 16, kk, lane);
        }
        // prefetch next A-fragments while the WMMAs run
        v16bf a0n, a1n;
        if (kk + 32 < EMB) {
            a0n = load_a_f32cvt(X, EMB, rowbase,      kk + 32, lane);
            a1n = load_a_f32cvt(X, EMB, rowbase + 16, kk + 32, lane);
        }

#pragma unroll
        for (int u = 0; u < 6; ++u) {
            acc0[u] = wmma_bf16(a0, bfr[u], acc0[u]);
            acc1[u] = wmma_bf16(a1, bfr[u], acc1[u]);
        }
        a0 = a0n;
        a1 = a1n;
    }

    const int n   = lane & 15;
    const int mhi = (lane >> 4) * 8;
    const int b    = rowbase >> 10;
    const int srow = rowbase & (SEQ - 1);
#pragma unroll
    for (int r = 0; r < 2; ++r) {
        const v8f* acc = r ? acc1 : acc0;
#pragma unroll
        for (int u = 0; u < 6; ++u) {
            const int t    = tbase + u;
            const int colg = (t & 3) * 16 + n;
            const float* bias = (t < 4) ? bq : (t < 8) ? bk : bv;
            const float bias_v = bias[colg];
#pragma unroll
            for (int v = 0; v < 8; ++v) {
                const int m   = r * 16 + mhi + v;
                const float val = acc[u][v] + bias_v;
                if (t < 4) {
                    Qbf[(size_t)(rowbase + m) * HD + colg] = (__bf16)val;
                } else if (t < 8) {
                    Kbf[(size_t)(rowbase + m) * HD + colg] = (__bf16)val;
                } else {
                    Vt[(size_t)b * HD * SEQ + (size_t)colg * SEQ + (srow + m)] = (__bf16)val;
                }
            }
        }
    }
}

// ---------------------------------------------------------------------------
// Kernel 2: flash attention.
// Grid (S/64, B) x 128 threads (4 waves). Each wave owns 16 query rows,
// streams keys in chunks of 32. S^T = K * Q^T, online softmax, H^T = V^T*P^T.
// ---------------------------------------------------------------------------
__global__ __launch_bounds__(128, 1) void attn_kernel(const __bf16* __restrict__ Qbf,
                                                      const __bf16* __restrict__ Kbf,
                                                      const __bf16* __restrict__ Vt,
                                                      const unsigned char* __restrict__ pmask,
                                                      float* __restrict__ H) {
    const int lane = threadIdx.x & 31;
    const int wave = threadIdx.x >> 5;
    const int b  = blockIdx.y;
    const int q0 = blockIdx.x * 64 + wave * 16;

    const __bf16* Qb = Qbf + (size_t)b * SEQ * HD;
    const __bf16* Kb = Kbf + (size_t)b * SEQ * HD;
    const __bf16* Vb = Vt  + (size_t)b * HD * SEQ;
    const unsigned char* pmb = pmask + (size_t)b * SEQ;

    // Q^T B-fragments (query in lanes, head-dim as K), live across the loop.
    v16bf qt0 = load_b_bf16(Qb, HD, q0, 0,  lane);
    v16bf qt1 = load_b_bf16(Qb, HD, q0, 32, lane);

    const int   n    = lane & 15;          // query column within tile
    const int   mhi  = (lane >> 4) * 8;    // key row sub-offset
    const int   qi   = q0 + n;
    const float scale = 0.125f;            // 1/sqrt(64)
    const bool  hihalf = (lane >= 16);

    float runmax = -__builtin_inff();
    float runsum = 0.0f;
    v8f acc[4] = {};

    for (int j0 = 0; j0 < SEQ; j0 += 32) {
        // --- preload all K fragments, then two 16x16 S^T score tiles ---
        v16bf ka = load_a_bf16(Kb, HD, j0,      0,  lane);
        v16bf kb = load_a_bf16(Kb, HD, j0,      32, lane);
        v16bf kc = load_a_bf16(Kb, HD, j0 + 16, 0,  lane);
        v16bf kd = load_a_bf16(Kb, HD, j0 + 16, 32, lane);
        // padding mask: two aligned 8-byte loads instead of 16 byte loads
        const unsigned long long pmA = *(const unsigned long long*)(pmb + j0 + mhi);
        const unsigned long long pmB = *(const unsigned long long*)(pmb + j0 + 16 + mhi);

        v8f s0 = {}, s1 = {};
        s0 = wmma_bf16(ka, qt0, s0);
        s0 = wmma_bf16(kb, qt1, s0);
        s1 = wmma_bf16(kc, qt0, s1);
        s1 = wmma_bf16(kd, qt1, s1);

        // --- issue V^T fragment loads now; they overlap the softmax VALU ---
        v16bf vfr[4];
#pragma unroll
        for (int t = 0; t < 4; ++t) vfr[t] = load_a_bf16(Vb, SEQ, t * 16, j0, lane);

        // --- mask + scale, per-lane partial max ---
        float p0[8], p1[8];
        float mloc = -__builtin_inff();
#pragma unroll
        for (int v = 0; v < 8; ++v) {
            int kj = j0 + mhi + v;
            float sv = s0[v] * scale;
            bool bad = (((pmA >> (8 * v)) & 0xffull) != 0) || (qi >= NIMG && kj > qi);
            sv = bad ? -__builtin_inff() : sv;
            p0[v] = sv; mloc = fmaxf(mloc, sv);

            kj += 16;
            sv = s1[v] * scale;
            bad = (((pmB >> (8 * v)) & 0xffull) != 0) || (qi >= NIMG && kj > qi);
            sv = bad ? -__builtin_inff() : sv;
            p1[v] = sv; mloc = fmaxf(mloc, sv);
        }
        // combine the two half-wave key subsets for this query
        mloc = fmaxf(mloc, __shfl_xor(mloc, 16, 32));

        const float newmax = fmaxf(runmax, mloc);
        const float base   = (newmax == -__builtin_inff()) ? 0.0f : newmax;
        const float corr   = __expf(runmax - base);   // runmax=-inf -> 0

        float lsum = 0.0f;
#pragma unroll
        for (int v = 0; v < 8; ++v) {
            p0[v] = __expf(p0[v] - base);
            p1[v] = __expf(p1[v] - base);
            lsum += p0[v] + p1[v];
        }
        lsum += __shfl_xor(lsum, 16, 32);
        runsum = runsum * corr + lsum;
        runmax = newmax;

        // --- build P^T B-fragment (32 keys x 16 queries) from the lane's own p0/p1 ---
        float x0[8], x1[8];
#pragma unroll
        for (int v = 0; v < 8; ++v) {
            x0[v] = __shfl_xor(p0[v], 16, 32);
            x1[v] = __shfl_xor(p1[v], 16, 32);
        }
        v16bf pfrag;
#pragma unroll
        for (int e = 0; e < 8; ++e) {
            pfrag[e]     = (__bf16)p0[e];
            pfrag[e + 8] = (__bf16)p1[e];
        }

        // --- rescale accumulators, then H^T += V^T * P^T (4 d-tiles) ---
#pragma unroll
        for (int t = 0; t < 4; ++t) {
#pragma unroll
            for (int v = 0; v < 8; ++v) acc[t][v] *= corr;
            acc[t] = wmma_bf16(vfr[t], pfrag, acc[t]);
        }
    }

    const float inv = (runsum > 0.0f) ? (1.0f / runsum) : 0.0f;
    float* Hb = H + (size_t)b * SEQ * HD;
#pragma unroll
    for (int t = 0; t < 4; ++t) {
#pragma unroll
        for (int v = 0; v < 8; ++v) {
            const int d = t * 16 + mhi + v;
            Hb[(size_t)qi * HD + d] = acc[t][v] * inv;
        }
    }
}

// ---------------------------------------------------------------------------
// Launcher
// ---------------------------------------------------------------------------
extern "C" void kernel_launch(void* const* d_in, const int* in_sizes, int n_in,
                              void* d_out, int out_size, void* d_ws, size_t ws_size,
                              hipStream_t stream) {
    const float* X  = (const float*)d_in[0];
    const float* Wq = (const float*)d_in[1];
    const float* bq = (const float*)d_in[2];
    const float* Wk = (const float*)d_in[3];
    const float* bk = (const float*)d_in[4];
    const float* Wv = (const float*)d_in[5];
    const float* bv = (const float*)d_in[6];
    // d_in[7] = input_sequence_length (unused by the reference math)
    const unsigned char* pmask = (const unsigned char*)d_in[8];
    float* out = (float*)d_out;

    char* ws = (char*)d_ws;
    __bf16* Qbf = (__bf16*)(ws);                      // 4 MiB
    __bf16* Kbf = (__bf16*)(ws + (4u << 20));         // 4 MiB
    __bf16* Vt  = (__bf16*)(ws + (8u << 20));         // 4 MiB
    __bf16* Wt  = (__bf16*)(ws + (12u << 20));        // 288 KiB

    const int wtotal = 3 * EMB * HD;
    prep_weights<<<(wtotal + 255) / 256, 256, 0, stream>>>(Wq, Wk, Wv, Wt);
    proj_kernel<<<dim3((BATCH * SEQ) / 128), dim3(256), 0, stream>>>(
        X, Wt, bq, bk, bv, Qbf, Kbf, Vt);
    attn_kernel<<<dim3(SEQ / 64, BATCH), dim3(128), 0, stream>>>(
        Qbf, Kbf, Vt, pmask, out);
}
